// SpaceFormerLite_49134425866853
// MI455X (gfx1250) — matrix operand order repaired and checked by reading, not compiled, not run on hardware
//
#include <hip/hip_runtime.h>
#include <hip/hip_bf16.h>
#include <math.h>

#define NROWS 8192
#define DDIM  512
#define NW32  (NROWS / 32)                 /* 256 mask words per row */
#define SCALE 0.044194173824159216f        /* 1/sqrt(512) */

typedef __bf16 bf16_t;
typedef __bf16 v16bf __attribute__((ext_vector_type(16)));
typedef __bf16 v8bf  __attribute__((ext_vector_type(8)));
typedef float  v8f   __attribute__((ext_vector_type(8)));
typedef float  v4f   __attribute__((ext_vector_type(4)));
typedef __bf16 v4bf  __attribute__((ext_vector_type(4)));

union BF16x16 { v16bf v; v8bf h[2]; bf16_t e[16]; };

__device__ __forceinline__ bf16_t f2bf(float f) { return (bf16_t)f; }

// ---------------- bulk f32 -> bf16 conversion (x, masked_x) ----------------
__global__ void k_cvt_bf16(const float* __restrict__ in, bf16_t* __restrict__ out,
                           long long n4) {
  const long long i = (long long)blockIdx.x * blockDim.x + threadIdx.x;
  if (i >= n4) return;
  v4f f = *reinterpret_cast<const v4f*>(in + i * 4);
  v4bf b;
#pragma unroll
  for (int e = 0; e < 4; ++e) b[e] = f2bf(f[e]);
  *reinterpret_cast<v4bf*>(out + i * 4) = b;
}

// ---------------- pack adjacency bools into 32-bit masks ----------------
__global__ void k_packadj(const unsigned char* __restrict__ adj,
                          unsigned int* __restrict__ adjb) {
  const long long wi = (long long)blockIdx.x * blockDim.x + threadIdx.x; // word index
  if (wi >= (long long)NROWS * NW32) return;
  const unsigned int* a32 =
      reinterpret_cast<const unsigned int*>(adj + wi * 32); // 32 bools = 8 u32
  unsigned int word = 0;
#pragma unroll
  for (int q = 0; q < 8; ++q) {
    const unsigned int v = a32[q];
#pragma unroll
    for (int b = 0; b < 4; ++b)
      if ((v >> (8 * b)) & 0xffu) word |= 1u << (q * 4 + b);
  }
  adjb[wi] = word;
}

// ---------------- bf16 transpose of a 512x512 f32 weight: Wt[j][i] = W[i][j] ----------------
__global__ void k_transpose_wt(const float* __restrict__ W, bf16_t* __restrict__ Wt) {
  const int t = blockIdx.x * blockDim.x + threadIdx.x;
  if (t >= DDIM * DDIM) return;
  const int j = t >> 9, i = t & 511;        // output-coalesced
  Wt[j * DDIM + i] = f2bf(W[i * DDIM + j]);
}

// ---------------- column mean: x_bar[c] = mean_n x[n,c] ----------------
__global__ void k_colmean(const float* __restrict__ x, float* __restrict__ xbar) {
  int c = blockIdx.x * blockDim.x + threadIdx.x;
  if (c >= DDIM) return;
  float s = 0.f;
  for (int n = 0; n < NROWS; ++n) s += x[n * DDIM + c];
  xbar[c] = s * (1.0f / NROWS);
}

// ---------------- gK = x_bar@Wgk ; gV = x_bar@Wgv + bgv ----------------
__global__ void k_gvec(const float* __restrict__ xbar, const float* __restrict__ Wgk,
                       const float* __restrict__ Wgv, const float* __restrict__ bgv,
                       float* __restrict__ gK, float* __restrict__ gV) {
  int j = blockIdx.x * blockDim.x + threadIdx.x;
  if (j >= DDIM) return;
  float sk = 0.f, sv = 0.f;
  for (int i = 0; i < DDIM; ++i) {
    float xb = xbar[i];
    sk += xb * Wgk[i * DDIM + j];
    sv += xb * Wgv[i * DDIM + j];
  }
  gK[j] = sk; gV[j] = sv + bgv[j];
}

// ---------------- wq = Wgq @ gK  (so gs = masked_x . wq) ----------------
__global__ void k_wq(const float* __restrict__ Wgq, const float* __restrict__ gK,
                     float* __restrict__ wq) {
  int i = blockIdx.x * blockDim.x + threadIdx.x;
  if (i >= DDIM) return;
  float s = 0.f;
  for (int j = 0; j < DDIM; ++j) s += Wgq[i * DDIM + j] * gK[j];
  wq[i] = s;
}

// ------------- per-row scalars: gss = scale*(mx.wq), es = mx.Wes -------------
__global__ void k_rowscalar(const float* __restrict__ mx, const float* __restrict__ wq,
                            const float* __restrict__ Wes,
                            float* __restrict__ gss, float* __restrict__ esr) {
  __shared__ float s1[256], s2[256];
  const int r = blockIdx.x, t = threadIdx.x;
  float a = mx[r * DDIM + t] * wq[t] + mx[r * DDIM + 256 + t] * wq[256 + t];
  float b = mx[r * DDIM + t] * Wes[t] + mx[r * DDIM + 256 + t] * Wes[256 + t];
  s1[t] = a; s2[t] = b;
  __syncthreads();
  for (int off = 128; off > 0; off >>= 1) {
    if (t < off) { s1[t] += s1[t + off]; s2[t] += s2[t + off]; }
    __syncthreads();
  }
  if (t == 0) { gss[r] = s1[0] * SCALE; esr[r] = s2[0]; }
}

// ---------------- WMMA GEMM: out = A[N,512] @ W (+bias); A bf16, B bf16 Wt[out][in] ----------------
// 256 threads = 8 waves; wave w owns output tile rows [bx*16,+16), cols [by*128+w*16,+16)
__global__ __launch_bounds__(256) void k_gemm_bf16(
    const bf16_t* __restrict__ A, const bf16_t* __restrict__ Wt,
    const float* __restrict__ bias,
    bf16_t* __restrict__ out_bf, float* __restrict__ out_f,
    bf16_t* __restrict__ out_tr) {
  const int w    = threadIdx.x >> 5;
  const int lane = threadIdx.x & 31;
  const int half = lane >> 4;
  const int lr   = lane & 15;
  const int row  = blockIdx.x * 16 + lr;                 // A-fragment row (M = lane%16)
  const int col  = blockIdx.y * 128 + w * 16 + lr;       // B-fragment col (N = lane%16)

  v8f acc = {};
  const bf16_t* ap = A + (long long)row * DDIM;
  const bf16_t* bp = Wt + (long long)col * DDIM;
  for (int kk = 0; kk < DDIM; kk += 32) {
    BF16x16 a;
    a.h[0] = *reinterpret_cast<const v8bf*>(ap + kk + half * 16);      // K = kk+half*16+e
    a.h[1] = *reinterpret_cast<const v8bf*>(ap + kk + half * 16 + 8);  // K = kk+half*16+8+e
    v16bf b = *reinterpret_cast<const v16bf*>(bp + kk + half * 16);    // K = kk+half*16+e
    acc = __builtin_amdgcn_wmma_f32_16x16x32_bf16(false, a.v, false, b,
                                                  (short)0, acc, false, false);
  }
  const float bv = bias ? bias[col] : 0.f;
#pragma unroll
  for (int v = 0; v < 8; ++v) {
    const int orow = blockIdx.x * 16 + v + 8 * half;     // C layout: M = v + 8*(lane/16)
    const float val = acc[v] + bv;
    if (out_bf) out_bf[(long long)orow * DDIM + col] = f2bf(val);
    if (out_f)  out_f [(long long)orow * DDIM + col] = val;
    if (out_tr) out_tr[(long long)col * NROWS + orow] = f2bf(val); // V stored [col][key]
  }
}

// ---------------- flash attention with joint (local+global+essential) softmax ----------------
// Block = 16 query rows, 8 waves. Wave w: K-slice [w*64,+64) for scores, out-cols [w*64,+64).
__global__ __launch_bounds__(256) void k_attn(
    const bf16_t* __restrict__ Qb, const bf16_t* __restrict__ Kb,
    const bf16_t* __restrict__ Vt,           // transposed: Vt[col*NROWS + key]
    const unsigned int* __restrict__ adjb,   // packed mask: adjb[row*NW32 + key/32]
    const float* __restrict__ gss, const float* __restrict__ esr,
    float* __restrict__ O, float* __restrict__ mOut, float* __restrict__ lOut) {
  __shared__ float  Spart[8 * 2 * 256];   // per-wave partial score fragments
  __shared__ float  Sfull[512];           // reduced scaled scores (fragment order)
  __shared__ bf16_t Ps[16 * 32];          // P tile, row-major [row][key]
  __shared__ float  mrow[16], lrow[16], fac[16];

  const int tid  = threadIdx.x;
  const int w    = tid >> 5, lane = tid & 31;
  const int half = lane >> 4, lr = lane & 15;
  const int qbase = blockIdx.x * 16;

  if (tid < 16) {  // joint softmax init: global + essential logits join the running stats
    const float g = gss[qbase + tid], e = esr[qbase + tid];
    const float m0 = fmaxf(g, e);
    mrow[tid] = m0;
    lrow[tid] = __expf(g - m0) + __expf(e - m0);
  }

  // Q fragments for this wave's 64-wide K slice (loop invariant -> registers)
  BF16x16 qa[2];
  {
    const bf16_t* qp = Qb + (long long)(qbase + lr) * DDIM + w * 64;
#pragma unroll
    for (int f = 0; f < 2; ++f) {
      qa[f].h[0] = *reinterpret_cast<const v8bf*>(qp + f * 32 + half * 16);
      qa[f].h[1] = *reinterpret_cast<const v8bf*>(qp + f * 32 + half * 16 + 8);
    }
  }

  v8f Cacc[4] = {};  // wave's 16x64 f32 output accumulator (4 C fragments)
  __syncthreads();

  for (int c = 0; c < NROWS / 32; ++c) {
    const int keyb = c * 32;
    if (c + 1 < NROWS / 32) {  // gfx1250 global_prefetch_b8 for next chunk
      __builtin_prefetch(Kb + (long long)(keyb + 32 + lr) * DDIM + w * 64, 0, 1);
      __builtin_prefetch(Vt + (long long)(w * 64 + lr) * NROWS + keyb + 32, 0, 1);
    }
    // 1) partial S over this wave's K slice: two 16x16 tiles (keys t*16..)
#pragma unroll
    for (int t = 0; t < 2; ++t) {
      v8f st = {};
      const int key = keyb + t * 16 + lr;
#pragma unroll
      for (int f = 0; f < 2; ++f) {
        v16bf kb = *reinterpret_cast<const v16bf*>(
            Kb + (long long)key * DDIM + w * 64 + f * 32 + half * 16);
        st = __builtin_amdgcn_wmma_f32_16x16x32_bf16(false, qa[f].v, false, kb,
                                                     (short)0, st, false, false);
      }
      float* sp = &Spart[(w * 2 + t) * 256 + lane * 8];
#pragma unroll
      for (int v = 0; v < 8; ++v) sp[v] = st[v];
    }
    __syncthreads();

    // 2) reduce 8 wave partials + scale (mask applied in step 3 from bitmask)
#pragma unroll
    for (int rr = 0; rr < 2; ++rr) {
      const int pp = tid + rr * 256;
      const int t = pp >> 8, p = pp & 255;
      float s = 0.f;
#pragma unroll
      for (int ww = 0; ww < 8; ++ww) s += Spart[(ww * 2 + t) * 256 + p];
      Sfull[pp] = s * SCALE;
    }
    __syncthreads();

    // 3) one lane per query row: mask (1 u32), online max/sum update, emit P tile
    if (tid < 16) {
      const int r = tid;
      const unsigned int mword = adjb[(long long)(qbase + r) * NW32 + (keyb >> 5)];
      float vals[32];
#pragma unroll
      for (int k2 = 0; k2 < 32; ++k2) {
        const int t = k2 >> 4, n = k2 & 15;
        const int L2 = n + 16 * (r >> 3);
        const float raw = Sfull[t * 256 + L2 * 8 + (r & 7)];
        vals[k2] = ((mword >> k2) & 1u) ? raw : -1e9f;
      }
      float cmax = vals[0];
#pragma unroll
      for (int k2 = 1; k2 < 32; ++k2) cmax = fmaxf(cmax, vals[k2]);
      const float mold = mrow[r];
      const float mnew = fmaxf(mold, cmax);
      const float corr = __expf(mold - mnew);
      float csum = 0.f;
#pragma unroll
      for (int k2 = 0; k2 < 32; ++k2) {
        const float pe = __expf(vals[k2] - mnew);
        csum += pe;
        Ps[r * 32 + k2] = f2bf(pe);
      }
      lrow[r] = lrow[r] * corr + csum;
      mrow[r] = mnew;
      fac[r] = corr;
    }
    __syncthreads();

    // 4) rescale accumulators, P @ V into wave's 64 output columns
    BF16x16 pa;
    pa.h[0] = *reinterpret_cast<const v8bf*>(&Ps[lr * 32 + half * 16]);
    pa.h[1] = *reinterpret_cast<const v8bf*>(&Ps[lr * 32 + half * 16 + 8]);
    float fv[8];
#pragma unroll
    for (int v = 0; v < 8; ++v) fv[v] = fac[v + 8 * half];
#pragma unroll
    for (int j = 0; j < 4; ++j) {
#pragma unroll
      for (int v = 0; v < 8; ++v) Cacc[j][v] *= fv[v];
      const int col = w * 64 + j * 16 + lr;
      // B fragment: contiguous 16 bf16 (K_local = half*16 + e) from transposed V
      v16bf vb = *reinterpret_cast<const v16bf*>(
          Vt + (long long)col * NROWS + keyb + half * 16);
      Cacc[j] = __builtin_amdgcn_wmma_f32_16x16x32_bf16(false, pa.v, false, vb,
                                                        (short)0, Cacc[j], false, false);
    }
  }

  // write unnormalized O (normalized in epilogue), plus per-row stats
#pragma unroll
  for (int j = 0; j < 4; ++j) {
    const int col = w * 64 + j * 16 + lr;
#pragma unroll
    for (int v = 0; v < 8; ++v)
      O[(long long)(qbase + v + 8 * half) * DDIM + col] = Cacc[j][v];
  }
  if (tid < 16) { mOut[qbase + tid] = mrow[tid]; lOut[qbase + tid] = lrow[tid]; }
}

// ---------------- epilogue: normalize, add global/essential terms, max with masked_x ----------------
__global__ void k_epilogue(const float* __restrict__ mstat, const float* __restrict__ lstat,
                           const float* __restrict__ gss, const float* __restrict__ esr,
                           const float* __restrict__ gV, const float* __restrict__ eV,
                           const float* __restrict__ mx, float* __restrict__ out) {
  const long long idx = (long long)blockIdx.x * blockDim.x + threadIdx.x;
  if (idx >= (long long)NROWS * DDIM) return;
  const int r = (int)(idx / DDIM), c2 = (int)(idx % DDIM);
  const float m = mstat[r], inv = 1.f / lstat[r];
  const float ge = __expf(gss[r] - m), ee = __expf(esr[r] - m);
  const float res = out[idx] * inv + ge * inv * gV[c2] + ee * inv * eV[idx];
  out[idx] = fmaxf(res, mx[idx]);
}

extern "C" void kernel_launch(void* const* d_in, const int* in_sizes, int n_in,
                              void* d_out, int out_size, void* d_ws, size_t ws_size,
                              hipStream_t stream) {
  const unsigned char* adj = (const unsigned char*)d_in[0];
  const float* x   = (const float*)d_in[1];
  const float* mx  = (const float*)d_in[2];
  const float* Wlq = (const float*)d_in[3];
  const float* Wlk = (const float*)d_in[4];
  const float* Wlv = (const float*)d_in[5];
  const float* blv = (const float*)d_in[6];
  const float* Wgq = (const float*)d_in[7];
  const float* Wgk = (const float*)d_in[8];
  const float* Wgv = (const float*)d_in[9];
  const float* bgv = (const float*)d_in[10];
  const float* Wes = (const float*)d_in[11];
  const float* Wev = (const float*)d_in[12];
  const float* bev = (const float*)d_in[13];
  float* out = (float*)d_out;

  char* ws = (char*)d_ws;
  size_t off = 0;
  auto alloc = [&](size_t bytes) -> void* {
    void* p = ws + off;
    off = (off + bytes + 255) & ~(size_t)255;
    return p;
  };
  bf16_t* xb   = (bf16_t*)alloc((size_t)NROWS * DDIM * sizeof(bf16_t));
  bf16_t* mxb  = (bf16_t*)alloc((size_t)NROWS * DDIM * sizeof(bf16_t));
  bf16_t* lQb  = (bf16_t*)alloc((size_t)NROWS * DDIM * sizeof(bf16_t));
  bf16_t* lKb  = (bf16_t*)alloc((size_t)NROWS * DDIM * sizeof(bf16_t));
  bf16_t* Vtb  = (bf16_t*)alloc((size_t)NROWS * DDIM * sizeof(bf16_t)); // [col][key]
  float*  eVf  = (float*) alloc((size_t)NROWS * DDIM * sizeof(float));
  unsigned int* adjb = (unsigned int*)alloc((size_t)NROWS * NW32 * sizeof(unsigned int));
  bf16_t* Wtb  = (bf16_t*)alloc((size_t)DDIM * DDIM * sizeof(bf16_t)); // reused per GEMM
  float* xbar  = (float*)alloc(DDIM * sizeof(float));
  float* gK    = (float*)alloc(DDIM * sizeof(float));
  float* gV    = (float*)alloc(DDIM * sizeof(float));
  float* wq    = (float*)alloc(DDIM * sizeof(float));
  float* gss   = (float*)alloc(NROWS * sizeof(float));
  float* esr   = (float*)alloc(NROWS * sizeof(float));
  float* mst   = (float*)alloc(NROWS * sizeof(float));
  float* lst   = (float*)alloc(NROWS * sizeof(float));

  const dim3 bl(256);
  const long long n4 = (long long)NROWS * DDIM / 4;
  k_cvt_bf16<<<dim3((unsigned)((n4 + 255) / 256)), bl, 0, stream>>>(x, xb, n4);
  k_cvt_bf16<<<dim3((unsigned)((n4 + 255) / 256)), bl, 0, stream>>>(mx, mxb, n4);
  k_packadj<<<dim3((NROWS * NW32) / 256), bl, 0, stream>>>(adj, adjb);
  k_colmean<<<dim3(2), bl, 0, stream>>>(x, xbar);

  const dim3 ggrid(NROWS / 16, DDIM / 128);
  const dim3 tgrid((DDIM * DDIM) / 256);
  // sequential reuse of Wtb is safe: all launches are stream-ordered
  k_transpose_wt<<<tgrid, bl, 0, stream>>>(Wlq, Wtb);
  k_gemm_bf16<<<ggrid, bl, 0, stream>>>(mxb, Wtb, nullptr, lQb, nullptr, nullptr);
  k_transpose_wt<<<tgrid, bl, 0, stream>>>(Wlk, Wtb);
  k_gemm_bf16<<<ggrid, bl, 0, stream>>>(xb,  Wtb, nullptr, lKb, nullptr, nullptr);
  k_transpose_wt<<<tgrid, bl, 0, stream>>>(Wlv, Wtb);
  k_gemm_bf16<<<ggrid, bl, 0, stream>>>(xb,  Wtb, blv,     nullptr, nullptr, Vtb);
  k_transpose_wt<<<tgrid, bl, 0, stream>>>(Wev, Wtb);
  k_gemm_bf16<<<ggrid, bl, 0, stream>>>(mxb, Wtb, bev,     nullptr, eVf, nullptr);

  k_gvec<<<dim3(2), bl, 0, stream>>>(xbar, Wgk, Wgv, bgv, gK, gV);
  k_wq<<<dim3(2), bl, 0, stream>>>(Wgq, gK, wq);
  k_rowscalar<<<dim3(NROWS), bl, 0, stream>>>(mx, wq, Wes, gss, esr);

  k_attn<<<dim3(NROWS / 16), bl, 0, stream>>>(lQb, lKb, Vtb, adjb, gss, esr, out, mst, lst);
  k_epilogue<<<dim3((NROWS * DDIM) / 256), bl, 0, stream>>>(mst, lst, gss, esr, gV, eVf, mx, out);
}
